// MultiheadAttention_1357209665608
// MI455X (gfx1250) — matrix operand order repaired and checked by reading, not compiled, not run on hardware
//
#include <hip/hip_runtime.h>

#define NBATCH 4
#define NHEADS 16
#define SEQ    2048
#define EMBED  1024
#define DH     64
#define QTILES (SEQ / 16)   // 128

typedef __attribute__((ext_vector_type(16))) _Float16 v16h;
typedef __attribute__((ext_vector_type(8)))  float    v8f;
typedef __attribute__((ext_vector_type(4)))  float    v4f;

__device__ __forceinline__ v8f wmma_f16(v16h a, v16h b, v8f c) {
  // D = A(16x32) * B(32x16) + C, f32 accumulate
  return __builtin_amdgcn_wmma_f32_16x16x32_f16(
      /*neg_a=*/false, a, /*neg_b=*/false, b,
      /*c_mod=*/(short)0, c, /*reuse_a=*/false, /*reuse_b=*/false);
}

// ---------------------------------------------------------------------------
// Pass 1: per-row softmax stats (m, l) via WMMA QK^T.
// Per-lane online accumulation; single cross-lane (m,l) merge at the end.
// One wave per (n, h, 16-row query tile). 4 waves per block.
// ---------------------------------------------------------------------------
__global__ __launch_bounds__(128) void attn_stats_kernel(
    const float* __restrict__ q, const float* __restrict__ kmat,
    float* __restrict__ m_ws, float* __restrict__ l_ws)
{
  const int lane = threadIdx.x & 31;
  const int wid  = blockIdx.x * 4 + (threadIdx.x >> 5);
  const int qt = wid & (QTILES - 1);
  const int nh = wid >> 7;           // wid / QTILES
  const int h  = nh & (NHEADS - 1);
  const int n  = nh >> 4;

  const int mrow = lane & 15;
  const int hi   = lane >> 4;
  const float scale = 0.125f;        // 1/sqrt(64)

  // Q tile (16 x 64) -> two A operands (k=32 each), A-layout
  v16h aq0, aq1;
  {
    const size_t qbase = ((size_t)(n * SEQ) + qt * 16 + mrow) * EMBED + h * DH;
#pragma unroll
    for (int i = 0; i < 16; ++i) {
      const int kk = (hi ? 16 : 0) + i;
      aq0[i] = (_Float16)q[qbase + kk];
      aq1[i] = (_Float16)q[qbase + 32 + kk];
    }
  }

  // per-lane partial stats over the key slots this lane sees (col = mrow mod 16)
  float mrun[8], lrun[8];
#pragma unroll
  for (int vv = 0; vv < 8; ++vv) { mrun[vv] = -3.0e38f; lrun[vv] = 0.0f; }

  for (int kt = 0; kt <= qt; ++kt) {
    const size_t kbase =
        ((size_t)(n * SEQ) + kt * 16 + mrow) * EMBED + h * DH + (hi ? 16 : 0);
    if (kt < qt) __builtin_prefetch(&kmat[kbase + EMBED * 16], 0, 1);
    v16h bk0, bk1;
#pragma unroll
    for (int i = 0; i < 16; ++i) {
      bk0[i] = (_Float16)kmat[kbase + i];
      bk1[i] = (_Float16)kmat[kbase + 32 + i];
    }

    v8f c = {};
    c = wmma_f16(aq0, bk0, c);
    c = wmma_f16(aq1, bk1, c);

#pragma unroll
    for (int vv = 0; vv < 8; ++vv) {
      const int row = vv + (hi << 3);
      float x = c[vv] * scale;
      if (kt == qt && mrow > row) x = -3.0e38f;   // causal mask, diagonal tile
      const float mnew = fmaxf(mrun[vv], x);
      lrun[vv] = lrun[vv] * __expf(mrun[vv] - mnew) + __expf(x - mnew);
      mrun[vv] = mnew;
    }
  }

  // merge the 16 per-lane partials of each row (associative (m,l) merge)
#pragma unroll
  for (int vv = 0; vv < 8; ++vv) {
    float m = mrun[vv], l = lrun[vv];
#pragma unroll
    for (int off = 1; off < 16; off <<= 1) {
      const float m2 = __shfl_xor(m, off);
      const float l2 = __shfl_xor(l, off);
      const float mo = fmaxf(m, m2);
      l = l * __expf(m - mo) + l2 * __expf(m2 - mo);
      m = mo;
    }
    mrun[vv] = m; lrun[vv] = l;
  }

  if (mrow == 0) {
#pragma unroll
    for (int vv = 0; vv < 8; ++vv) {
      const int row = qt * 16 + vv + (hi << 3);
      const size_t idx = ((size_t)(n * NHEADS) + h) * SEQ + row;
      m_ws[idx] = mrun[vv];
      l_ws[idx] = lrun[vv];
    }
  }
}

// ---------------------------------------------------------------------------
// Pass 2: recompute scores, write normalized attention weights (coalesced,
// non-temporal), and fuse y += P * V via WMMA (P staged through LDS as f32,
// re-read both for the NT global write and as the f16 A-operand).
// ---------------------------------------------------------------------------
__global__ __launch_bounds__(128) void attn_out_kernel(
    const float* __restrict__ q, const float* __restrict__ kmat,
    const float* __restrict__ vmat,
    const float* __restrict__ m_ws, const float* __restrict__ l_ws,
    float* __restrict__ y, float* __restrict__ attn)
{
  __shared__ float lds_w[4][16 * 32];   // per-wave 16x32 P staging (2 KB each)

  const int lane  = threadIdx.x & 31;
  const int wslot = threadIdx.x >> 5;
  const int wid   = blockIdx.x * 4 + wslot;
  const int qt = wid & (QTILES - 1);
  const int nh = wid >> 7;
  const int h  = nh & (NHEADS - 1);
  const int n  = nh >> 4;

  const int mrow = lane & 15;
  const int hi   = lane >> 4;
  const float scale = 0.125f;

  // Q tile A operands
  v16h aq0, aq1;
  {
    const size_t qbase = ((size_t)(n * SEQ) + qt * 16 + mrow) * EMBED + h * DH;
#pragma unroll
    for (int i = 0; i < 16; ++i) {
      const int kk = (hi ? 16 : 0) + i;
      aq0[i] = (_Float16)q[qbase + kk];
      aq1[i] = (_Float16)q[qbase + 32 + kk];
    }
  }

  // softmax stats for the 8 rows this lane owns (replicated across 16 lanes)
  float m8[8], linv8[8];
#pragma unroll
  for (int vv = 0; vv < 8; ++vv) {
    const int row = qt * 16 + vv + (hi << 3);
    const size_t idx = ((size_t)(n * NHEADS) + h) * SEQ + row;
    m8[vv] = m_ws[idx];
    linv8[vv] = 1.0f / l_ws[idx];
  }

  v8f yacc[4] = {};   // 4 column tiles of dh_v = 64

  const size_t attn_row_base =
      (((size_t)(n * NHEADS) + h) * SEQ + qt * 16) * (size_t)SEQ;

  const int jmax = qt >> 1;   // 32-key blocks covering keys <= qt*16+15
  for (int j = 0; j <= jmax; ++j) {
#pragma unroll
    for (int s = 0; s < 2; ++s) {
      const int kt = 2 * j + s;
      const size_t kbase =
          ((size_t)(n * SEQ) + kt * 16 + mrow) * EMBED + h * DH + (hi ? 16 : 0);
      v16h bk0, bk1;
#pragma unroll
      for (int i = 0; i < 16; ++i) {
        bk0[i] = (_Float16)kmat[kbase + i];
        bk1[i] = (_Float16)kmat[kbase + 32 + i];
      }
      v8f c = {};
      c = wmma_f16(aq0, bk0, c);
      c = wmma_f16(aq1, bk1, c);

#pragma unroll
      for (int vv = 0; vv < 8; ++vv) {
        const int row = vv + (hi << 3);          // local row 0..15
        const int key = kt * 16 + mrow;          // global key index
        float x = c[vv] * scale;
        if (key > qt * 16 + row) x = -3.0e38f;   // causal
        const float p = __expf(x - m8[vv]) * linv8[vv];
        lds_w[wslot][row * 32 + s * 16 + mrow] = p;
      }
    }
    asm volatile("s_wait_dscnt 0" ::: "memory");

    // coalesced non-temporal attn writes: 16 rows x 32 cols per block
    {
      const int wr_row = lane >> 1;              // 0..15
      const int wr_col = (lane & 1) * 16;        // 0 or 16
      float* dst = &attn[attn_row_base + (size_t)wr_row * SEQ + j * 32 + wr_col];
#pragma unroll
      for (int qd = 0; qd < 4; ++qd) {
        const v4f val = *(const v4f*)&lds_w[wslot][wr_row * 32 + wr_col + qd * 4];
        __builtin_nontemporal_store(val, (v4f*)(dst + qd * 4));
      }
    }

    // gather P (16x32) in WMMA A-layout from LDS, converting to f16
    v16h ap;
#pragma unroll
    for (int i = 0; i < 16; ++i) {
      const int kk = (hi ? 16 : 0) + i;
      ap[i] = (_Float16)lds_w[wslot][mrow * 32 + kk];
    }

    // V block as B operands, accumulate 4 output column tiles (L2-resident)
#pragma unroll
    for (int t = 0; t < 4; ++t) {
      v16h bv;
#pragma unroll
      for (int i = 0; i < 16; ++i) {
        const int key = j * 32 + (hi ? 16 : 0) + i;
        bv[i] = (_Float16)vmat[((size_t)(n * SEQ) + key) * EMBED +
                               h * DH + t * 16 + mrow];
      }
      yacc[t] = wmma_f16(ap, bv, yacc[t]);
    }
  }

  // zero-fill the fully masked upper-triangle attn columns (d_out is poisoned)
  const int zstart = (jmax + 1) * 32;
  for (int col = zstart + mrow * 4; col < SEQ; col += 64) {
#pragma unroll
    for (int vv = 0; vv < 8; ++vv) {
      const int row = vv + (hi << 3);
      const v4f z = {0.f, 0.f, 0.f, 0.f};
      __builtin_nontemporal_store(
          z, (v4f*)&attn[attn_row_base + (size_t)row * SEQ + col]);
    }
  }

  // write y
#pragma unroll
  for (int t = 0; t < 4; ++t) {
#pragma unroll
    for (int vv = 0; vv < 8; ++vv) {
      const int row = qt * 16 + vv + (hi << 3);
      __builtin_nontemporal_store(
          yacc[t][vv],
          &y[((size_t)(n * SEQ) + row) * EMBED + h * DH + t * 16 + mrow]);
    }
  }
}

// ---------------------------------------------------------------------------
extern "C" void kernel_launch(void* const* d_in, const int* in_sizes, int n_in,
                              void* d_out, int out_size, void* d_ws, size_t ws_size,
                              hipStream_t stream) {
  const float* q = (const float*)d_in[0];
  const float* k = (const float*)d_in[1];
  const float* v = (const float*)d_in[2];
  // d_in[3] = is_causal (always 1 in this workload; kernels assume causal)

  float* y    = (float*)d_out;
  float* attn = (float*)d_out + (size_t)NBATCH * SEQ * EMBED;  // outputs: (y, attn)

  float* m_ws = (float*)d_ws;                                  // N*H*L floats
  float* l_ws = m_ws + (size_t)NBATCH * NHEADS * SEQ;          // N*H*L floats

  const int total_waves = NBATCH * NHEADS * QTILES;            // 8192
  dim3 block(128);                                             // 4 waves (wave32)
  dim3 grid(total_waves / 4);

  attn_stats_kernel<<<grid, block, 0, stream>>>(q, k, m_ws, l_ws);
  attn_out_kernel<<<grid, block, 0, stream>>>(q, k, v, m_ws, l_ws, y, attn);
}
